// SimpleRNN_56032143344055
// MI455X (gfx1250) — compile-verified, matrix operand order checked by reading, not run on hardware
//
#include <hip/hip_runtime.h>
#include <cstdint>
#include <cstddef>

// Problem dims (fixed by the reference)
#define B_  128
#define T_  1024
#define I_  512
#define H_  1024
#define O_  256

typedef __attribute__((ext_vector_type(16))) __bf16 v16bf;
typedef __attribute__((ext_vector_type(8)))  float  v8f;

union FragU { uint4 u[2]; v16bf v; };

// Load one 16-lane-striped bf16 WMMA operand fragment.
// ISA 16-bit A layout (CDNA5 ISA §7.12.2): lane L holds row m = L&15;
// hi = L>>4 selects K = hi*8+{0..7} (VGPR0-3) and K = 16+hi*8+{0..7} (VGPR4-7)
// -> two contiguous 16-byte chunks per lane. Same pattern serves the B operand
// when B is provided transposed (row n = L&15 of B^T).
static __device__ __forceinline__ v16bf load_frag_bf16(const __bf16* rowptr, int hi) {
  FragU t;
  t.u[0] = *reinterpret_cast<const uint4*>(rowptr + hi * 8);
  t.u[1] = *reinterpret_cast<const uint4*>(rowptr + 16 + hi * 8);
  return t.v;
}

// Same fragment, but source row is f32: load 16 floats, convert (RNE) to bf16.
static __device__ __forceinline__ v16bf load_frag_f32_cvt(const float* rowptr, int hi) {
  const float4* p = reinterpret_cast<const float4*>(rowptr + hi * 8);
  const float4* q = reinterpret_cast<const float4*>(rowptr + 16 + hi * 8);
  float4 f0 = p[0], f1 = p[1], f2 = q[0], f3 = q[1];
  v16bf v;
  v[0]  = (__bf16)f0.x; v[1]  = (__bf16)f0.y; v[2]  = (__bf16)f0.z; v[3]  = (__bf16)f0.w;
  v[4]  = (__bf16)f1.x; v[5]  = (__bf16)f1.y; v[6]  = (__bf16)f1.z; v[7]  = (__bf16)f1.w;
  v[8]  = (__bf16)f2.x; v[9]  = (__bf16)f2.y; v[10] = (__bf16)f2.z; v[11] = (__bf16)f2.w;
  v[12] = (__bf16)f3.x; v[13] = (__bf16)f3.y; v[14] = (__bf16)f3.z; v[15] = (__bf16)f3.w;
  return v;
}

// out[C][R] (bf16) = transpose(in[R][C] f32). Writes coalesced, reads strided;
// only used on the small weight matrices (<= 4 MB).
__global__ void transpose_cvt_bf16(const float* __restrict__ in, __bf16* __restrict__ out,
                                   int R, int C) {
  long idx = (long)blockIdx.x * blockDim.x + threadIdx.x;
  long total = (long)R * C;
  if (idx >= total) return;
  int r = (int)(idx % R);
  int c = (int)(idx / R);
  out[idx] = (__bf16)in[(long)r * C + c];  // idx == c*R + r
}

// xp[BT][H] (bf16) = x[BT][I] @ kernel[I][H] + bias.  kT = kernel^T [H][I] bf16.
// One wave computes a 16(M) x 64(N) tile; 8 waves / block.
__global__ void gemm_xp_kernel(const float* __restrict__ x, const __bf16* __restrict__ kT,
                               const float* __restrict__ bias, __bf16* __restrict__ xp) {
  const int lane = threadIdx.x & 31;
  const int wave = threadIdx.x >> 5;
  const int w    = blockIdx.x * 8 + wave;    // 0 .. 131071
  const int bm   = (w >> 4) << 4;            // M tile base (8192 tiles)
  const int bn   = (w & 15) << 6;            // N base (16 groups of 64)
  const int hi   = lane >> 4;
  const int r    = lane & 15;

  const float*  arow = x  + (size_t)(bm + r) * I_;
  const __bf16* brow = kT + (size_t)(bn + r) * I_;

  v8f acc0 = {}, acc1 = {}, acc2 = {}, acc3 = {};
#pragma unroll 4
  for (int kb = 0; kb < I_; kb += 32) {
    v16bf a  = load_frag_f32_cvt(arow + kb, hi);
    v16bf b0 = load_frag_bf16(brow + kb,             hi);
    v16bf b1 = load_frag_bf16(brow + 16 * I_ + kb,   hi);
    v16bf b2 = load_frag_bf16(brow + 32 * I_ + kb,   hi);
    v16bf b3 = load_frag_bf16(brow + 48 * I_ + kb,   hi);
    acc0 = __builtin_amdgcn_wmma_f32_16x16x32_bf16(false, a, false, b0, (short)0, acc0, false, false);
    acc1 = __builtin_amdgcn_wmma_f32_16x16x32_bf16(false, a, false, b1, (short)0, acc1, false, false);
    acc2 = __builtin_amdgcn_wmma_f32_16x16x32_bf16(false, a, false, b2, (short)0, acc2, false, false);
    acc3 = __builtin_amdgcn_wmma_f32_16x16x32_bf16(false, a, false, b3, (short)0, acc3, false, false);
  }

  v8f accs[4] = {acc0, acc1, acc2, acc3};
#pragma unroll
  for (int nt = 0; nt < 4; ++nt) {
    const int col = bn + nt * 16 + r;        // D layout: col n = lane&15
    const float bv = bias[col];
#pragma unroll
    for (int q = 0; q < 8; ++q) {            // D layout: row = q + hi*8
      const int row = bm + q + hi * 8;
      xp[(size_t)row * H_ + col] = (__bf16)(accs[nt][q] + bv);
    }
  }
}

// Persistent recurrent scan: 64 blocks x 256 threads (512 waves == 512 16x16 tiles
// of the [128 x 1024] hidden state). W^T rows for this block's 4 N-tiles are staged
// into LDS (64 rows x 2 KB = 128 KB) once and re-read every timestep. Device-wide
// sync via a monotonic arrival counter (counter must be zeroed before launch).
__global__ void rnn_scan_kernel(const __bf16* __restrict__ xp, const __bf16* __restrict__ Wt,
                                __bf16* __restrict__ hb0, __bf16* __restrict__ hb1,
                                unsigned* __restrict__ barrier_cnt) {
  extern __shared__ char smem_raw[];
  __bf16* sW = reinterpret_cast<__bf16*>(smem_raw);   // [64][H_]

  const int lane     = threadIdx.x & 31;
  const int wave     = threadIdx.x >> 5;
  const int hi       = lane >> 4;
  const int r        = lane & 15;
  const int mt_local = wave >> 2;                       // 0..1
  const int nt_local = wave & 3;                        // 0..3
  const int bm       = ((((int)blockIdx.x >> 4) << 1) + mt_local) << 4;   // 0..112
  const int bn_block = ((int)blockIdx.x & 15) << 6;                        // 0..960
  const int bn       = bn_block + (nt_local << 4);

  // Stage Wt[bn_block .. bn_block+63][*] into LDS.
  {
    const uint4* src = reinterpret_cast<const uint4*>(Wt + (size_t)bn_block * H_);
    uint4* dst = reinterpret_cast<uint4*>(sW);
    const int n16 = 64 * H_ / 8;                        // 8192 uint4
    for (int i = threadIdx.x; i < n16; i += blockDim.x) dst[i] = src[i];
  }
  __syncthreads();

  __bf16* bufs[2] = { hb0, hb1 };
  const __bf16* brow = sW + (size_t)((nt_local << 4) + r) * H_;
  const unsigned nb = gridDim.x;

  for (int t = 0; t < T_; ++t) {
    const __bf16* hc = bufs[t & 1];
    __bf16*       hn = bufs[(t + 1) & 1];
    const __bf16* arow = hc + (size_t)(bm + r) * H_;

    if (t + 1 < T_)  // warm next step's xp line (global_prefetch_b8)
      __builtin_prefetch(&xp[((size_t)(bm + r) * T_ + (t + 1)) * H_ + bn], 0, 0);

    v8f acc = {};
#pragma unroll 4
    for (int kb = 0; kb < H_; kb += 32) {
      v16bf a = load_frag_bf16(arow + kb, hi);
      v16bf b = load_frag_bf16(brow + kb, hi);
      acc = __builtin_amdgcn_wmma_f32_16x16x32_bf16(false, a, false, b, (short)0, acc, false, false);
    }

    const int col = bn + r;
#pragma unroll
    for (int q = 0; q < 8; ++q) {
      const int row = bm + q + hi * 8;
      float xv = (float)xp[((size_t)row * T_ + t) * H_ + col];
      float hv = tanhf(acc[q] + xv);
      hn[(size_t)row * H_ + col] = (__bf16)hv;
    }

    // ---- device-wide barrier (release -> arrive -> spin -> acquire) ----
    __threadfence();
    __syncthreads();
    if (threadIdx.x == 0) {
      atomicAdd(barrier_cnt, 1u);
      const unsigned want = (unsigned)(t + 1) * nb;
      while (__hip_atomic_load(barrier_cnt, __ATOMIC_RELAXED, __HIP_MEMORY_SCOPE_AGENT) < want)
        __builtin_amdgcn_s_sleep(2);
    }
    __syncthreads();
    __threadfence();
  }
}

// out[128][256] f32 = h_last[128][1024] @ fc_w[1024][256] + fc_b.  fwT = fc_w^T.
__global__ void fc_kernel(const __bf16* __restrict__ h, const __bf16* __restrict__ fwT,
                          const float* __restrict__ fc_b, float* __restrict__ out) {
  const int lane = threadIdx.x & 31;
  const int wave = threadIdx.x >> 5;
  const int w    = blockIdx.x * 8 + wave;   // 0..127
  const int bm   = (w >> 4) << 4;           // 8 M tiles
  const int bn   = (w & 15) << 4;           // 16 N tiles
  const int hi   = lane >> 4;
  const int r    = lane & 15;

  const __bf16* arow = h   + (size_t)(bm + r) * H_;
  const __bf16* brow = fwT + (size_t)(bn + r) * H_;

  v8f acc = {};
#pragma unroll 4
  for (int kb = 0; kb < H_; kb += 32) {
    v16bf a = load_frag_bf16(arow + kb, hi);
    v16bf b = load_frag_bf16(brow + kb, hi);
    acc = __builtin_amdgcn_wmma_f32_16x16x32_bf16(false, a, false, b, (short)0, acc, false, false);
  }

  const int col = bn + r;
  const float bv = fc_b[col];
#pragma unroll
  for (int q = 0; q < 8; ++q)
    out[(size_t)(bm + q + hi * 8) * O_ + col] = acc[q] + bv;
}

extern "C" void kernel_launch(void* const* d_in, const int* in_sizes, int n_in,
                              void* d_out, int out_size, void* d_ws, size_t ws_size,
                              hipStream_t stream) {
  const float* x      = (const float*)d_in[0];   // [128,1024,512]
  const float* kernel = (const float*)d_in[1];   // [512,1024]
  const float* rec    = (const float*)d_in[2];   // [1024,1024]
  const float* bias   = (const float*)d_in[3];   // [1024]
  const float* fc_w   = (const float*)d_in[4];   // [1024,256]
  const float* fc_b   = (const float*)d_in[5];   // [256]
  float* out = (float*)d_out;                    // [128,256]

  unsigned char* ws = (unsigned char*)d_ws;
  // Workspace layout (256-byte aligned):
  __bf16*   kT   = (__bf16*)(ws + 0);                      //  1 MB  kernel^T  [H][I]
  __bf16*   Wt   = (__bf16*)(ws + (1u << 20));             //  2 MB  rec^T     [H][H]
  __bf16*   fwT  = (__bf16*)(ws + (3u << 20));             //  0.5MB fc_w^T    [O][H]
  __bf16*   h0   = (__bf16*)(ws + 3670016u);               //  256KB hidden buf 0
  __bf16*   h1   = (__bf16*)(ws + 3932160u);               //  256KB hidden buf 1
  unsigned* cnt  = (unsigned*)(ws + 4194304u);             //  barrier counter
  __bf16*   xp   = (__bf16*)(ws + 4194560u);               //  256MB xp [B*T][H] bf16

  // 1) Weight transpose + f32->bf16 conversion.
  transpose_cvt_bf16<<<(I_ * H_ + 255) / 256, 256, 0, stream>>>(kernel, kT, I_, H_);
  transpose_cvt_bf16<<<(H_ * H_ + 255) / 256, 256, 0, stream>>>(rec, Wt, H_, H_);
  transpose_cvt_bf16<<<(H_ * O_ + 255) / 256, 256, 0, stream>>>(fc_w, fwT, H_, O_);

  // 2) Zero initial hidden state and barrier counter (every call: graph-replay safe).
  hipMemsetAsync(h0, 0, (size_t)B_ * H_ * sizeof(__bf16), stream);
  hipMemsetAsync(cnt, 0, 256, stream);

  // 3) Input projection for all timesteps (throughput WMMA GEMM).
  gemm_xp_kernel<<<(B_ * T_ / 16) * (H_ / 64) / 8, 256, 0, stream>>>(x, kT, bias, xp);

  // 4) Persistent recurrent scan: 64 co-resident blocks, grid barrier per step.
  //    After t = 1023 (odd) the final hidden state lives in h0.
  rnn_scan_kernel<<<64, 256, 64 * H_ * sizeof(__bf16), stream>>>(xp, Wt, h0, h1, cnt);

  // 5) Output dense layer.
  fc_kernel<<<16, 256, 0, stream>>>(h0, fwT, fc_b, out);
}
